// MultiHeadAttention_27616639713824
// MI455X (gfx1250) — matrix operand order repaired and checked by reading, not compiled, not run on hardware
//
#include <hip/hip_runtime.h>
#include <hip/hip_bf16.h>

typedef __attribute__((ext_vector_type(8)))  __bf16 v8bf;
typedef __attribute__((ext_vector_type(16))) __bf16 v16bf;
typedef __attribute__((ext_vector_type(8)))  float  v8f;

#define S_TOK 35
#define SP    48
#define EMB   512
#define NHEAD 8
#define HDIM  64

// ---- LDS layout (bytes) ----
// Qs  : [48][512] bf16  (reused as attn output before out-projection)
// Ks  : [48][512] bf16
// Vt  : [512][64] bf16  (V transposed, token dim padded 48->64 with zeros)
// Xs  : [48][512] bf16  (aliased by Sc after QKV gemm)
// Sc  : [8][48][48] f32 (raw scores)
// Ps  : [8][48][64] bf16 (softmax probs, col-padded with zeros)
#define OFF_Q   0
#define OFF_K   49152
#define OFF_VT  98304
#define OFF_X   163840
#define OFF_SC  163840
#define OFF_P   (163840 + 73728)
#define LDS_BYTES (OFF_P + 49152)   // 286720 B <= 320KB WGP LDS

static __device__ __forceinline__ unsigned short f2bf(float f) {
  unsigned u = __builtin_bit_cast(unsigned, f);
  unsigned r = u + 0x7FFFu + ((u >> 16) & 1u);   // round-to-nearest-even
  return (unsigned short)(r >> 16);
}

// A fragment (16x32 bf16, M x K): lane holds row m, 16 contiguous K {kb..kb+15}
static __device__ __forceinline__ v16bf ldA(const unsigned short* base, int ld, int row, int kb) {
  const v8bf lo = *(const v8bf*)(base + row * ld + kb);
  const v8bf hi = *(const v8bf*)(base + row * ld + kb + 8);
  return __builtin_shufflevector(lo, hi, 0,1,2,3,4,5,6,7,8,9,10,11,12,13,14,15);
}

// B fragment (32x16 bf16, K x N) loaded from Bt row-major: lane holds col n, 16 contiguous K
static __device__ __forceinline__ v16bf ldBt(const unsigned short* p) {
  const v8bf lo = *(const v8bf*)(p);
  const v8bf hi = *(const v8bf*)(p + 8);
  return __builtin_shufflevector(lo, hi, 0,1,2,3,4,5,6,7,8,9,10,11,12,13,14,15);
}

static __device__ __forceinline__ v8f wmma_bf16(v16bf a, v16bf b, v8f c) {
  return __builtin_amdgcn_wmma_f32_16x16x32_bf16(false, a, false, b, (short)0, c, false, false);
}

// ---------- pre-kernel: fp32 weights -> bf16 (Wcat = [Wq;Wk;Wv] rows, Wo separate) ----------
__global__ void prep_weights(const float* __restrict__ Wq, const float* __restrict__ Wk,
                             const float* __restrict__ Wv, const float* __restrict__ Wo,
                             unsigned short* __restrict__ wcat, unsigned short* __restrict__ wobf) {
  int i = blockIdx.x * blockDim.x + threadIdx.x;
  if (i < EMB * EMB) {
    wcat[i]                 = f2bf(Wq[i]);
    wcat[i + EMB * EMB]     = f2bf(Wk[i]);
    wcat[i + 2 * EMB * EMB] = f2bf(Wv[i]);
    wobf[i]                 = f2bf(Wo[i]);
  }
}

// ---------- fused attention: one block per batch element ----------
__global__ __launch_bounds__(256) void fused_attn(
    const float* __restrict__ x, const unsigned short* __restrict__ wcat,
    const unsigned short* __restrict__ wobf, const float* __restrict__ bo,
    const float* __restrict__ tscale, const float* __restrict__ tmask,
    float* __restrict__ out)
{
  extern __shared__ char smem[];
  unsigned short* Qs = (unsigned short*)(smem + OFF_Q);
  unsigned short* Ks = (unsigned short*)(smem + OFF_K);
  unsigned short* Vt = (unsigned short*)(smem + OFF_VT);
  unsigned short* Xs = (unsigned short*)(smem + OFF_X);
  float*          Sc = (float*)(smem + OFF_SC);
  unsigned short* Ps = (unsigned short*)(smem + OFF_P);

  const int tid  = threadIdx.x;
  const int lane = tid & 31;
  const int wave = tid >> 5;
  const int l15  = lane & 15;
  const int hi16 = (lane >> 4) & 1;
  const int b    = blockIdx.x;
  const float* xb = x + (size_t)b * (S_TOK * EMB);

  // ---- stage 1: zero Vt pad cols (t=48..63) and load x -> Xs bf16 (rows 35..47 zero) ----
  {
    unsigned* vtz = (unsigned*)Vt;
    for (int i = tid; i < EMB * 8; i += 256) {       // 512 rows x 8 uint pairs
      int d = i >> 3, j = i & 7;
      vtz[d * 32 + 24 + j] = 0u;
    }
    unsigned* xu = (unsigned*)Xs;
    for (int i = tid; i < (SP * EMB) / 2; i += 256) {
      int e0 = i * 2;
      int row = e0 >> 9;
      int col = e0 & 511;
      unsigned u = 0u;
      if (row < S_TOK) {
        float f0 = xb[row * EMB + col];
        float f1 = xb[row * EMB + col + 1];
        u = (unsigned)f2bf(f0) | ((unsigned)f2bf(f1) << 16);
      }
      xu[i] = u;
    }
  }
  __syncthreads();

  // ---- stage 2: QKV = Xs @ Wcat^T ; 288 (16x16) output tiles over 8 waves ----
  for (int t = wave; t < 288; t += 8) {
    const int m0 = (t % 3) * 16;
    const int ng = t / 3;                       // 0..95 : output N-tile across [Q|K|V]
    const int ncol = ng * 16 + l15;             // global output column for this lane
    const unsigned short* wrow = wcat + (size_t)ncol * EMB;
    __builtin_prefetch(wrow, 0, 0);
    v8f c = {0.f, 0.f, 0.f, 0.f, 0.f, 0.f, 0.f, 0.f};
    for (int kk = 0; kk < EMB; kk += 32) {
      v16bf a  = ldA(Xs, EMB, m0 + l15, kk + hi16 * 16);
      v16bf bm = ldBt(wrow + kk + hi16 * 16);
      c = wmma_bf16(a, bm, c);
    }
    if (ng < 64) {                              // Q or K : row-major bf16
      unsigned short* dst = (ng < 32) ? Qs : Ks;
      const int col = (ng & 31) * 16 + l15;
      #pragma unroll
      for (int r = 0; r < 8; ++r) {
        int row = m0 + r + hi16 * 8;
        dst[row * EMB + col] = f2bf(c[r]);
      }
    } else {                                    // V : store transposed Vt[d][t]
      const int d = (ng - 64) * 16 + l15;
      #pragma unroll
      for (int r = 0; r < 8; ++r) {
        int tt = m0 + r + hi16 * 8;
        Vt[d * 64 + tt] = f2bf(c[r]);
      }
    }
  }
  __syncthreads();

  // ---- stage 3: scores = Q @ K^T, one head per wave (Sc aliases Xs: x is dead) ----
  {
    const int h = wave;
    float* sch = Sc + h * SP * SP;
    #pragma unroll
    for (int mt = 0; mt < 3; ++mt) {
      #pragma unroll
      for (int nt = 0; nt < 3; ++nt) {
        v8f c = {0.f, 0.f, 0.f, 0.f, 0.f, 0.f, 0.f, 0.f};
        #pragma unroll
        for (int kk = 0; kk < HDIM; kk += 32) {
          v16bf a  = ldA(Qs, EMB, mt * 16 + l15, h * HDIM + kk + hi16 * 16);
          v16bf bm = ldBt(Ks + (nt * 16 + l15) * EMB + h * HDIM + kk + hi16 * 16);
          c = wmma_bf16(a, bm, c);
        }
        #pragma unroll
        for (int r = 0; r < 8; ++r) {
          int row = mt * 16 + r + hi16 * 8;
          sch[row * SP + nt * 16 + l15] = c[r];
        }
      }
    }
  }
  __syncthreads();

  // ---- stage 4: softmax rows (fp32) + topology bias; write probs bf16, zero padding ----
  {
    unsigned* pu = (unsigned*)Ps;               // zero pad rows 35..47 of Ps
    for (int i = tid; i < NHEAD * 13 * 32; i += 256) {
      int h = i / (13 * 32);
      int rem = i % (13 * 32);
      int row = 35 + rem / 32;
      int j = rem & 31;
      pu[h * SP * 32 + row * 32 + j] = 0u;
    }
    const float inv = 0.125f;                   // 1/sqrt(64)
    for (int r = tid; r < NHEAD * S_TOK; r += 256) {
      const int h = r / S_TOK, s = r % S_TOK;
      const float sc = tscale[h];
      const float* srow = Sc + h * SP * SP + s * SP;
      const float* mrow = tmask + (h * S_TOK + s) * S_TOK;
      float vmax = -3.4e38f;
      for (int t2 = 0; t2 < S_TOK; ++t2)
        vmax = fmaxf(vmax, srow[t2] * inv + sc * mrow[t2]);
      float sum = 0.f;
      for (int t2 = 0; t2 < S_TOK; ++t2)
        sum += __expf(srow[t2] * inv + sc * mrow[t2] - vmax);
      const float rs = 1.f / sum;
      unsigned short* prow = Ps + h * SP * 64 + s * 64;
      for (int t2 = 0; t2 < S_TOK; ++t2)
        prow[t2] = f2bf(__expf(srow[t2] * inv + sc * mrow[t2] - vmax) * rs);
      for (int t2 = S_TOK; t2 < 64; ++t2) prow[t2] = 0;
    }
  }
  __syncthreads();

  // ---- stage 5: attn = P @ V (via Vt), one head per wave; write over Qs ----
  {
    const int h = wave;
    const unsigned short* Ph = Ps + h * SP * 64;
    #pragma unroll
    for (int mt = 0; mt < 3; ++mt) {
      #pragma unroll
      for (int nt = 0; nt < 4; ++nt) {
        v8f c = {0.f, 0.f, 0.f, 0.f, 0.f, 0.f, 0.f, 0.f};
        #pragma unroll
        for (int kk = 0; kk < 64; kk += 32) {
          v16bf a  = ldA(Ph, 64, mt * 16 + l15, kk + hi16 * 16);
          v16bf bm = ldBt(Vt + (h * HDIM + nt * 16 + l15) * 64 + kk + hi16 * 16);
          c = wmma_bf16(a, bm, c);
        }
        #pragma unroll
        for (int r = 0; r < 8; ++r) {
          int row = mt * 16 + r + hi16 * 8;
          Qs[row * EMB + h * HDIM + nt * 16 + l15] = f2bf(c[r]);
        }
      }
    }
  }
  __syncthreads();

  // ---- stage 6: out = attn @ Wo^T + bo, fp32 to HBM (rows < 35 only) ----
  float* outb = out + (size_t)b * (S_TOK * EMB);
  for (int t = wave; t < 96; t += 8) {
    const int m0 = (t % 3) * 16;
    const int nt = t / 3;                       // 0..31
    const int ncol = nt * 16 + l15;
    const unsigned short* wrow = wobf + (size_t)ncol * EMB;
    __builtin_prefetch(wrow, 0, 0);
    v8f c = {0.f, 0.f, 0.f, 0.f, 0.f, 0.f, 0.f, 0.f};
    for (int kk = 0; kk < EMB; kk += 32) {
      v16bf a  = ldA(Qs, EMB, m0 + l15, kk + hi16 * 16);
      v16bf bm = ldBt(wrow + kk + hi16 * 16);
      c = wmma_bf16(a, bm, c);
    }
    const float bias = bo[ncol];
    #pragma unroll
    for (int r = 0; r < 8; ++r) {
      int row = m0 + r + hi16 * 8;
      if (row < S_TOK) outb[row * EMB + ncol] = c[r] + bias;
    }
  }
}

extern "C" void kernel_launch(void* const* d_in, const int* in_sizes, int n_in,
                              void* d_out, int out_size, void* d_ws, size_t ws_size,
                              hipStream_t stream) {
  (void)in_sizes; (void)n_in; (void)out_size; (void)ws_size;
  const float* x  = (const float*)d_in[0];
  const float* Wq = (const float*)d_in[1];
  const float* Wk = (const float*)d_in[2];
  const float* Wv = (const float*)d_in[3];
  const float* Wo = (const float*)d_in[4];
  const float* bo = (const float*)d_in[5];
  const float* ts = (const float*)d_in[6];
  const float* tm = (const float*)d_in[7];

  unsigned short* wcat = (unsigned short*)d_ws;          // 1536*512 bf16
  unsigned short* wobf = wcat + 3 * EMB * EMB;           //  512*512 bf16

  (void)hipFuncSetAttribute((const void*)fused_attn,
                            hipFuncAttributeMaxDynamicSharedMemorySize, LDS_BYTES);

  prep_weights<<<(EMB * EMB + 255) / 256, 256, 0, stream>>>(Wq, Wk, Wv, Wo, wcat, wobf);
  fused_attn<<<4096, 256, LDS_BYTES, stream>>>(x, wcat, wobf, bo, ts, tm, (float*)d_out);
}
